// WordEmbCbowNet_27264452395031
// MI455X (gfx1250) — matrix operand order repaired and checked by reading, not compiled, not run on hardware
//
#include <hip/hip_runtime.h>

#define VOCAB   50257
#define EMBED   128
#define BATCH   4096
#define CTX     8
#define NTILES  3144            // ceil(50257/16)=3142, padded to multiple of 8
#define KTILES  4               // 128 / 32

typedef __attribute__((ext_vector_type(16))) __bf16 v16bf;
typedef __attribute__((ext_vector_type(8)))  float  v8f;

#if defined(__AMDGCN__) && __has_builtin(__builtin_amdgcn_global_load_async_to_lds_b128)
#define HAVE_ASYNC_LDS 1
#else
#define HAVE_ASYNC_LDS 0
#endif

#define GLOBAL_AS __attribute__((address_space(1)))
#define LDS_AS    __attribute__((address_space(3)))

// GCC-style vector (matches the builtin's declared parameter type) + AS-qualified ptrs
typedef int v4i __attribute__((vector_size(16)));
typedef GLOBAL_AS v4i* gv4ip;
typedef LDS_AS    v4i* lv4ip;

// LDS layout (in u16 units). A rows padded 128->136 elems; B lane chunks 16->20.
#define SA_STRIDE 136
#define SA_ELEMS  (128 * SA_STRIDE)       // 17408
#define SB_LANE   20
#define SB_FRAG   (32 * SB_LANE)          // 640
#define SB_ELEMS  (32 * SB_FRAG)          // 8 ntiles * 4 kt = 32 frags -> 20480
#define OFF_AHI   0
#define OFF_ALO   (SA_ELEMS)
#define OFF_BHI   (2 * SA_ELEMS)
#define OFF_BLO   (2 * SA_ELEMS + SB_ELEMS)
#define SMEM_U16  (2 * SA_ELEMS + 2 * SB_ELEMS)   // 75776 u16 = 151552 B

// ---------- bf16 split helpers (round-to-nearest-even) ----------
static __device__ __forceinline__ unsigned short f32_to_bf16_bits(float f) {
    unsigned u = __builtin_bit_cast(unsigned, f);
    unsigned r = u + 0x7FFFu + ((u >> 16) & 1u);
    return (unsigned short)(r >> 16);
}
static __device__ __forceinline__ float bf16_bits_to_f32(unsigned short h) {
    unsigned u = ((unsigned)h) << 16;
    return __builtin_bit_cast(float, u);
}
static __device__ __forceinline__ void split_bf16(float x, unsigned short& hi, unsigned short& lo) {
    hi = f32_to_bf16_bits(x);
    lo = f32_to_bf16_bits(x - bf16_bits_to_f32(hi));
}

// ---------- Stage 1: embedding bag + hi/lo split ----------
__global__ __launch_bounds__(EMBED) void embed_bag_split(
    const int* __restrict__ idx, const float* __restrict__ Wproj,
    unsigned short* __restrict__ Ahi, unsigned short* __restrict__ Alo)
{
    const int b = blockIdx.x;
    const int e = threadIdx.x;
    const int* row = idx + b * CTX;
    float s = 0.0f;
#pragma unroll
    for (int c = 0; c < CTX; ++c)
        s += Wproj[(long)row[c] * EMBED + e];
    unsigned short hi, lo;
    split_bf16(s, hi, lo);
    Ahi[(long)b * EMBED + e] = hi;
    Alo[(long)b * EMBED + e] = lo;
}

// ---------- Stage 2: pack W_pred into WMMA B-fragment layout, hi/lo split ----------
// Fragment f = ntile*KTILES + kt occupies 512 bf16; lane L owns 16 contiguous:
//   L < 16 : col = ntile*16 + L,      k = kt*32 + j       (j = 0..15)
//   L >= 16: col = ntile*16 + (L-16), k = kt*32 + 16 + j
__global__ __launch_bounds__(256) void pack_b_split(
    const float* __restrict__ Wpred,
    unsigned short* __restrict__ Bhi, unsigned short* __restrict__ Blo)
{
    const long t = (long)blockIdx.x * 256 + threadIdx.x;
    const long total = (long)NTILES * KTILES * 32 * 16;
    if (t >= total) return;
    const int j     = (int)(t & 15);
    const int lane  = (int)((t >> 4) & 31);
    const int kt    = (int)((t >> 9) & 3);
    const int ntile = (int)(t >> 11);
    const int n = ntile * 16 + (lane & 15);
    const int k = kt * 32 + ((lane >= 16) ? 16 : 0) + j;
    const float v = (n < VOCAB) ? Wpred[(long)k * VOCAB + n] : 0.0f;
    unsigned short hi, lo;
    split_bf16(v, hi, lo);
    Bhi[t] = hi;
    Blo[t] = lo;
}

// ---------- 16-byte global -> LDS copy (async on gfx1250, sync fallback) ----------
static __device__ __forceinline__ void cp16(unsigned short* lds, const unsigned short* g) {
#if HAVE_ASYNC_LDS
    __builtin_amdgcn_global_load_async_to_lds_b128(
        (gv4ip)(GLOBAL_AS void*)(void*)g,
        (lv4ip)(LDS_AS void*)lds,
        /*offset=*/0, /*cpol=*/0);
#else
    *(int4*)lds = *(const int4*)g;
#endif
}
static __device__ __forceinline__ void wait_async_then_barrier() {
#if HAVE_ASYNC_LDS
#if __has_builtin(__builtin_amdgcn_s_wait_asynccnt)
    __builtin_amdgcn_s_wait_asynccnt(0);
#else
    asm volatile("s_wait_asynccnt 0x0" ::: "memory");
#endif
#endif
    __syncthreads();
}

// ---------- fragment loads ----------
static __device__ __forceinline__ v16bf load_frag_b(const unsigned short* p) {   // 16 contiguous
    v16bf r;
    ((int4*)&r)[0] = *(const int4*)(p);
    ((int4*)&r)[1] = *(const int4*)(p + 8);
    return r;
}
static __device__ __forceinline__ v16bf load_frag_a(const unsigned short* p) {   // 16 contiguous
    v16bf r;
    ((int4*)&r)[0] = *(const int4*)(p);
    ((int4*)&r)[1] = *(const int4*)(p + 8);
    return r;
}
static __device__ __forceinline__ v8f wmma_bf16(v16bf a, v16bf b, v8f c) {
    return __builtin_amdgcn_wmma_f32_16x16x32_bf16(
        false, a, false, b, (short)0, c, false, false);
}

// ---------- Stage 3: GEMM via split-bf16 WMMA with LDS staging ----------
// WG = 256 threads = 8 waves, tile 128 rows x 128 cols (8 ntiles).
// wave: wm = wave>>2 (0..1, 64 rows each), wn = wave&3 (0..3, 2 ntiles each).
// Per wave: M=4 m-tiles, N=2 n-tiles, 4kt*2n*4m*3 = 96 WMMAs.
__global__ __launch_bounds__(256) void cbow_gemm_wmma(
    const unsigned short* __restrict__ Ahi, const unsigned short* __restrict__ Alo,
    const unsigned short* __restrict__ Bhi, const unsigned short* __restrict__ Blo,
    float* __restrict__ out)
{
    extern __shared__ unsigned short smem[];

    // ---- stage A (128x128 hi/lo) and B (32 frags hi/lo) into LDS ----
    const unsigned short* gAhi = Ahi + (size_t)blockIdx.y * (128 * EMBED);
    const unsigned short* gAlo = Alo + (size_t)blockIdx.y * (128 * EMBED);
    const unsigned short* gBhi = Bhi + (size_t)blockIdx.x * (8 * KTILES * 512);
    const unsigned short* gBlo = Blo + (size_t)blockIdx.x * (8 * KTILES * 512);
#pragma unroll
    for (int i = 0; i < 8; ++i) {
        const int c = threadIdx.x + i * 256;          // 0..2047 (16B chunks)
        const int arow = c >> 4, aj = c & 15;         // A: 16 chunks per 128-elem row
        const int aoff = arow * SA_STRIDE + aj * 8;
        cp16(smem + OFF_AHI + aoff, gAhi + c * 8);
        cp16(smem + OFF_ALO + aoff, gAlo + c * 8);
        const int blc = c >> 1, bj = c & 1;           // B: 2 chunks per 16-elem lane slot
        const int boff = blc * SB_LANE + bj * 8;
        cp16(smem + OFF_BHI + boff, gBhi + c * 8);
        cp16(smem + OFF_BLO + boff, gBlo + c * 8);
    }
    wait_async_then_barrier();

    // ---- compute ----
    const int lane = threadIdx.x & 31;
    const int wave = threadIdx.x >> 5;
    const int wm = wave >> 2;                         // 0..1
    const int wn = wave & 3;                          // 0..3
    const int ar = lane & 15;                         // A row within m-tile
    const int ak = (lane >> 4) * 16;                  // A K sub-offset per lane half

    v8f acc[4][2];
#pragma unroll
    for (int m = 0; m < 4; ++m)
#pragma unroll
        for (int n = 0; n < 2; ++n) acc[m][n] = (v8f){0,0,0,0,0,0,0,0};

#pragma unroll
    for (int kt = 0; kt < KTILES; ++kt) {
        v16bf ah[4], al[4];
#pragma unroll
        for (int m = 0; m < 4; ++m) {
            const int aoff = (wm * 64 + m * 16 + ar) * SA_STRIDE + kt * 32 + ak;
            ah[m] = load_frag_a(smem + OFF_AHI + aoff);
            al[m] = load_frag_a(smem + OFF_ALO + aoff);
        }
#pragma unroll
        for (int n = 0; n < 2; ++n) {
            const int f = (wn * 2 + n) * KTILES + kt;
            const int boff = f * SB_FRAG + lane * SB_LANE;
            const v16bf bh = load_frag_b(smem + OFF_BHI + boff);
            const v16bf bl = load_frag_b(smem + OFF_BLO + boff);
#pragma unroll
            for (int m = 0; m < 4; ++m) {
                acc[m][n] = wmma_bf16(ah[m], bh, acc[m][n]);   // hi*hi
                acc[m][n] = wmma_bf16(ah[m], bl, acc[m][n]);   // hi*lo
                acc[m][n] = wmma_bf16(al[m], bh, acc[m][n]);   // lo*hi
            }
        }
    }

    // ---- store: VGPR r -> (lanes 0-15: M=r) (lanes 16-31: M=8+r), N = lane&15 ----
    const int rhalf = (lane < 16) ? 0 : 8;
#pragma unroll
    for (int n = 0; n < 2; ++n) {
        const int col = blockIdx.x * 128 + (wn * 2 + n) * 16 + (lane & 15);
        if (col < VOCAB) {
#pragma unroll
            for (int m = 0; m < 4; ++m) {
                const int rbase = blockIdx.y * 128 + wm * 64 + m * 16 + rhalf;
#pragma unroll
                for (int r = 0; r < 8; ++r)
                    out[(long)(rbase + r) * VOCAB + col] = acc[m][n][r];
            }
        }
    }
}

extern "C" void kernel_launch(void* const* d_in, const int* in_sizes, int n_in,
                              void* d_out, int out_size, void* d_ws, size_t ws_size,
                              hipStream_t stream) {
    const int*   idx   = (const int*)d_in[0];
    const float* Wproj = (const float*)d_in[1];
    const float* Wpred = (const float*)d_in[2];
    float* out = (float*)d_out;

    char* ws = (char*)d_ws;
    const size_t A_BYTES = (size_t)BATCH * EMBED * 2;            // 1 MB each
    const size_t B_BYTES = (size_t)NTILES * KTILES * 512 * 2;    // ~12.88 MB each
    unsigned short* Ahi = (unsigned short*)(ws);
    unsigned short* Alo = (unsigned short*)(ws + A_BYTES);
    unsigned short* Bhi = (unsigned short*)(ws + 2 * A_BYTES);
    unsigned short* Blo = (unsigned short*)(ws + 2 * A_BYTES + B_BYTES);

    // Stage 1: embedding bag + split
    embed_bag_split<<<BATCH, EMBED, 0, stream>>>(idx, Wproj, Ahi, Alo);

    // Stage 2: pack B fragments (zero-padded tail tiles)
    const long b_elems = (long)NTILES * KTILES * 512;            // 6,438,912
    pack_b_split<<<(unsigned)((b_elems + 255) / 256), 256, 0, stream>>>(Wpred, Bhi, Blo);

    // Stage 3: WMMA GEMM with LDS staging (dynamic LDS = 151552 B, 2 WGs/WGP)
    dim3 grid(NTILES / 8, BATCH / 128);                          // 393 x 32
    cbow_gemm_wmma<<<grid, 256, SMEM_U16 * sizeof(unsigned short), stream>>>(
        Ahi, Alo, Bhi, Blo, out);
}
